// GNNAgent_14267881358066
// MI455X (gfx1250) — matrix operand order repaired and checked by reading, not compiled, not run on hardware
//
#include <hip/hip_runtime.h>

// Problem constants (from reference): G=512, N=128, F_IN=64, D=128, R=4, E=1572864, L=2
#define GG     512
#define NPG    128
#define FIN    64
#define DIM    128
#define RREL   4
#define NEDGE  1572864
#define NNODE  (GG * NPG)        // 65536
#define KCAT   (DIM + RREL*DIM)  // 640

typedef __attribute__((ext_vector_type(16))) __bf16 v16bf;
typedef __attribute__((ext_vector_type(8)))  float  v8f;

// Native fp32 -> bf16 conversion (compiler lowers to v_cvt*bf16* on gfx1250).
__device__ __forceinline__ __bf16 f2bf(float f) { return (__bf16)f; }

// Convert 8 consecutive fp32 (optionally scaled) into bf16 lanes [OFF, OFF+8).
template <int OFF, bool SCALE>
__device__ __forceinline__ void cvt8(v16bf& a, const float* p, float s) {
    const float4 u = ((const float4*)p)[0];
    const float4 w = ((const float4*)p)[1];
    float f0 = u.x, f1 = u.y, f2 = u.z, f3 = u.w;
    float f4 = w.x, f5 = w.y, f6 = w.z, f7 = w.w;
    if (SCALE) { f0 *= s; f1 *= s; f2 *= s; f3 *= s; f4 *= s; f5 *= s; f6 *= s; f7 *= s; }
    a[OFF + 0] = f2bf(f0); a[OFF + 1] = f2bf(f1);
    a[OFF + 2] = f2bf(f2); a[OFF + 3] = f2bf(f3);
    a[OFF + 4] = f2bf(f4); a[OFF + 5] = f2bf(f5);
    a[OFF + 6] = f2bf(f6); a[OFF + 7] = f2bf(f7);
}

// ---------------------------------------------------------------------------
// Weight prep: transpose + convert weights to bf16.
//   WtE  [128][64]   : WtE[j][k]      = W_emb[k][j]
//   WtL  [2][128][640]: WtL[l][j][k]  = (k<128) ? W_root[l][k][j]
//                                     : W_rel[l][(k-128)>>7][(k-128)&127][j]
// ---------------------------------------------------------------------------
__global__ __launch_bounds__(256) void prep_kernel(const float* __restrict__ W_emb,
                                                   const float* __restrict__ W_root,
                                                   const float* __restrict__ W_rel,
                                                   __bf16* __restrict__ WtE,
                                                   __bf16* __restrict__ WtL) {
    int i = blockIdx.x * 256 + threadIdx.x;
    const int nE = DIM * FIN;              // 8192
    const int nLper = DIM * KCAT;          // 81920
    if (i < nE) {
        int j = i >> 6, k = i & 63;
        WtE[i] = f2bf(W_emb[k * DIM + j]);
        return;
    }
    i -= nE;
    if (i >= 2 * nLper) return;
    int l   = i / nLper;
    int rem = i - l * nLper;
    int j   = rem / KCAT;
    int k   = rem - j * KCAT;
    float v;
    if (k < DIM) {
        v = W_root[((size_t)l * DIM + k) * DIM + j];
    } else {
        int r  = (k - DIM) >> 7;
        int kk = (k - DIM) & 127;
        v = W_rel[(((size_t)l * RREL + r) * DIM + kk) * DIM + j];
    }
    WtL[i] = f2bf(v);
}

// ---------------------------------------------------------------------------
// Edge scatter: one wave32 per edge. Gathers h[src] (L2-resident, 33.5 MB)
// and atomically accumulates into sum_h[(dst*R+rel)] (L2-resident, 134 MB),
// plus per-(dst,rel) edge count. unsafeAtomicAdd -> global_atomic_add_f32.
// ---------------------------------------------------------------------------
__global__ __launch_bounds__(256) void edge_kernel(const float* __restrict__ h,
                                                   const int* __restrict__ ei,
                                                   const int* __restrict__ ea,
                                                   float* __restrict__ sums,
                                                   float* __restrict__ cnt) {
    int wave = blockIdx.x * 8 + (threadIdx.x >> 5);
    int lane = threadIdx.x & 31;
    if (wave >= NEDGE) return;
    int src = ei[wave];
    int dst = ei[NEDGE + wave];
    int rel = ea[wave];
    const float4 v = *(const float4*)(h + (size_t)src * DIM + lane * 4);
    float* p = sums + ((size_t)dst * RREL + rel) * DIM + lane * 4;
    unsafeAtomicAdd(p + 0, v.x);
    unsafeAtomicAdd(p + 1, v.y);
    unsafeAtomicAdd(p + 2, v.z);
    unsafeAtomicAdd(p + 3, v.w);
    if (lane == 0) unsafeAtomicAdd(cnt + (size_t)dst * RREL + rel, 1.0f);
}

// ---------------------------------------------------------------------------
// WMMA GEMM: out[M=65536][128] = act( A[M][K] @ Wt^T + bias )
//   LAYER=false: A row = A0[row][0..K)                            (embed, K=64)
//   LAYER=true : A row = [ h[row][0..128), mean_r0, .., mean_r3 ] (K=640)
//                where mean_r = sum_h[row*R+r] * 1/max(cnt,1)
// K loop is structured as 128-wide regions so the A source pointer/scale is
// wave-uniform and compile-time after unrolling; region 0 (and embed) skips
// the scale multiplies entirely.
// Block: 256 thr = 8 waves; each wave: 16 rows x all 128 cols (8 C tiles).
// Fragments: A/B use contiguous K order (k = 16*hb + s), not the ISA table order (bf16 16x16x32, f32 accum).
// ---------------------------------------------------------------------------
template <int K, bool RELU, bool LAYER>
__global__ __launch_bounds__(256) void gemm_kernel(const float* __restrict__ A0,
                                                   const float* __restrict__ sums,
                                                   const float* __restrict__ cnt,
                                                   const __bf16* __restrict__ Wt,
                                                   const float* __restrict__ bias,
                                                   float* __restrict__ out) {
    const int lane = threadIdx.x & 31;
    const int wv   = threadIdx.x >> 5;
    const int m    = lane & 15;   // A row-in-tile / B column-in-tile
    const int hb   = lane >> 4;   // half selector
    const int row  = blockIdx.x * 128 + wv * 16 + m;   // node index (A row)

    float sc[RREL];
    if (LAYER) {
#pragma unroll
        for (int r = 0; r < RREL; ++r)
            sc[r] = 1.0f / fmaxf(cnt[(size_t)row * RREL + r], 1.0f);
    }

    v8f acc[8] = {};

    const int NREG = LAYER ? (K / DIM) : 1;   // 5 regions for layers, 1 for embed
    const int KREG = LAYER ? DIM : K;         // columns per region

#pragma unroll
    for (int reg = 0; reg < NREG; ++reg) {
        const float* base;
        float s;
        if (LAYER && reg > 0) {
            base = sums + ((size_t)row * RREL + (reg - 1)) * DIM;
            s = sc[reg - 1];
        } else {
            base = A0 + (size_t)row * (LAYER ? DIM : K);
            s = 1.0f;
        }
#pragma unroll
        for (int kk = 0; kk < KREG; kk += 32) {
            v16bf a;
            const float* pc = base + kk + hb * 16;
            if (LAYER && reg > 0) {
                cvt8<0, true>(a, pc, s);        // A frag K = kk + 16*hb + [0,8)
                cvt8<8, true>(a, pc + 8, s);    // A frag K = kk + 16*hb + 8 + [0,8)
            } else {
                cvt8<0, false>(a, pc, s);
                cvt8<8, false>(a, pc + 8, s);
            }
            const int kb = reg * DIM + kk;      // global K offset (reg==0 for embed)
#pragma unroll
            for (int t = 0; t < 8; ++t) {
                // B 32x16: lane n=m holds column n, 16 contiguous K at kb + 16*hb
                const v16bf b = *(const v16bf*)(Wt + (size_t)(t * 16 + m) * K + kb + hb * 16);
                acc[t] = __builtin_amdgcn_wmma_f32_16x16x32_bf16(
                    false, a, false, b, (short)0, acc[t], false, false);
            }
        }
    }

    // C layout: VGPR v -> M = hb*8 + v, N = t*16 + m
    const int obase = blockIdx.x * 128 + wv * 16 + hb * 8;
#pragma unroll
    for (int t = 0; t < 8; ++t) {
        const int col = t * 16 + m;
        const float bv = bias[col];
#pragma unroll
        for (int v = 0; v < 8; ++v) {
            float x = acc[t][v] + bv;
            if (RELU) x = fmaxf(x, 0.0f);
            out[(size_t)(obase + v) * DIM + col] = x;
        }
    }
}

// ---------------------------------------------------------------------------
// Per-graph max pool: out[g][d] = max_n h[g*128+n][d]
// ---------------------------------------------------------------------------
__global__ __launch_bounds__(256) void pool_kernel(const float* __restrict__ h,
                                                   float* __restrict__ out) {
    int idx = blockIdx.x * 256 + threadIdx.x;   // g*128 + d, 65536 total
    int g = idx >> 7, d = idx & 127;
    const float* p = h + (size_t)g * NPG * DIM + d;
    float mx = p[0];
#pragma unroll 4
    for (int n = 1; n < NPG; ++n) mx = fmaxf(mx, p[(size_t)n * DIM]);
    out[idx] = mx;
}

extern "C" void kernel_launch(void* const* d_in, const int* in_sizes, int n_in,
                              void* d_out, int out_size, void* d_ws, size_t ws_size,
                              hipStream_t stream) {
    const float* x      = (const float*)d_in[0];
    const int*   ei     = (const int*)d_in[1];
    const int*   ea     = (const int*)d_in[2];
    const float* W_emb  = (const float*)d_in[3];
    const float* b_emb  = (const float*)d_in[4];
    const float* W_root = (const float*)d_in[5];
    const float* W_rel  = (const float*)d_in[6];
    const float* bias   = (const float*)d_in[7];
    float* out = (float*)d_out;

    char* ws = (char*)d_ws;
    size_t off = 0;
    auto alloc = [&](size_t bytes) -> void* {
        off = (off + 255) & ~(size_t)255;
        void* p = ws + off;
        off += bytes;
        return p;
    };
    float*  h0   = (float*)alloc((size_t)NNODE * DIM * 4);          // 33.5 MB
    float*  h1   = (float*)alloc((size_t)NNODE * DIM * 4);          // 33.5 MB
    float*  sums = (float*)alloc((size_t)NNODE * RREL * DIM * 4);   // 134 MB
    float*  cnt  = (float*)alloc((size_t)NNODE * RREL * 4);         // 1 MB
    __bf16* WtE  = (__bf16*)alloc((size_t)DIM * FIN * 2);
    __bf16* WtL  = (__bf16*)alloc((size_t)2 * DIM * KCAT * 2);

    // 0) weight transpose + bf16 convert
    {
        int total = DIM * FIN + 2 * DIM * KCAT;   // 172032
        prep_kernel<<<(total + 255) / 256, 256, 0, stream>>>(W_emb, W_root, W_rel, WtE, WtL);
    }

    // 1) embedding: h0 = x @ W_emb + b_emb  (no relu)
    gemm_kernel<FIN, false, false><<<NNODE / 128, 256, 0, stream>>>(
        x, nullptr, nullptr, WtE, b_emb, h0);

    // 2) layer 0: aggregate h0, fused [root|rel] GEMM -> relu -> h1
    hipMemsetAsync(sums, 0, (size_t)NNODE * RREL * DIM * 4, stream);
    hipMemsetAsync(cnt,  0, (size_t)NNODE * RREL * 4, stream);
    edge_kernel<<<NEDGE / 8, 256, 0, stream>>>(h0, ei, ea, sums, cnt);
    gemm_kernel<KCAT, true, true><<<NNODE / 128, 256, 0, stream>>>(
        h0, sums, cnt, WtL, bias, h1);

    // 3) layer 1: h1 -> h0
    hipMemsetAsync(sums, 0, (size_t)NNODE * RREL * DIM * 4, stream);
    hipMemsetAsync(cnt,  0, (size_t)NNODE * RREL * 4, stream);
    edge_kernel<<<NEDGE / 8, 256, 0, stream>>>(h1, ei, ea, sums, cnt);
    gemm_kernel<KCAT, true, true><<<NNODE / 128, 256, 0, stream>>>(
        h1, sums, cnt, WtL + (size_t)DIM * KCAT, bias + DIM, h0);

    // 4) per-graph max pool -> out [512][128]
    pool_kernel<<<(GG * DIM) / 256, 256, 0, stream>>>(h0, out);
}
